// LSTM_28209345200271
// MI455X (gfx1250) — matrix operand order repaired and checked by reading, not compiled, not run on hardware
//
#include <hip/hip_runtime.h>
#include <hip/hip_fp16.h>

typedef __attribute__((ext_vector_type(16))) _Float16 v16h;
typedef __attribute__((ext_vector_type(8)))  _Float16 v8h;
typedef __attribute__((ext_vector_type(8)))  float    v8f;
typedef __attribute__((ext_vector_type(4)))  int      v4i;

typedef __attribute__((address_space(1))) v4i* gptr_v4i;
typedef __attribute__((address_space(3))) v4i* lptr_v4i;

#define T_  512
#define B_  64
#define I_  256
#define H_  512
#define G4  2048   // 4*H
#define NWG 16
#define HSL 32     // hidden units per WG
#define GSL 128    // gate columns per WG (4 gates * HSL)

// ---------------------------------------------------------------------------
// CDNA5 async copy to LDS (ASYNCcnt path), with portable fallback
// ---------------------------------------------------------------------------
__device__ __forceinline__ void async_copy_b128(_Float16* ldst, const _Float16* gsrc) {
#if __has_builtin(__builtin_amdgcn_global_load_async_to_lds_b128)
  __builtin_amdgcn_global_load_async_to_lds_b128(
      (gptr_v4i)(void*)(const_cast<_Float16*>(gsrc)),
      (lptr_v4i)(void*)ldst,
      0, 0);
#else
  *(v8h*)ldst = *(const v8h*)gsrc;
#endif
}

__device__ __forceinline__ void wait_async_zero() {
#if __has_builtin(__builtin_amdgcn_s_wait_asynccnt)
  __builtin_amdgcn_s_wait_asynccnt(0);
#elif __has_builtin(__builtin_amdgcn_global_load_async_to_lds_b128)
  asm volatile("s_wait_asynccnt 0" ::: "memory");
#endif
}

// ---------------------------------------------------------------------------
// WMMA helpers (16x16x32 f16 -> f32), operand layouts per ISA 7.12.2
// ---------------------------------------------------------------------------
__device__ __forceinline__ v16h cat16(v8h lo, v8h hi) {
  return __builtin_shufflevector(lo, hi, 0,1,2,3,4,5,6,7,8,9,10,11,12,13,14,15);
}

// A tile 16x32: p0 = &src[m0*ld + k0], row-major, ld in halves.
// lanes 0-15: row=lane,  K {0..7, 16..23}; lanes 16-31: row=lane-16, K {8..15, 24..31}
__device__ __forceinline__ v16h load_A(const _Float16* p0, int ld, int lane) {
  int m = lane & 15;
  int k = (lane >> 4) << 3;                // 0 or 8
  const _Float16* p = p0 + m * ld + k;
  v8h lo = *(const v8h*)p;                 // K = k .. k+7
  v8h hi = *(const v8h*)(p + 16);          // K = k+16 .. k+23
  return cat16(lo, hi);
}

// B tile 32x16 where B[k][n] = M[n0+n][k0+k] for row-major M (W or U).
// lanes 0-15: col=lane, K {0..7, 16..23}; lanes 16-31: col=lane-16, K {8..15, 24..31}
__device__ __forceinline__ v16h load_B(const _Float16* p0, int ld, int lane) {
  int n = lane & 15;
  int k = (lane >> 4) << 3;                // 0 or 8
  const _Float16* p = p0 + n * ld + k;
  v8h lo = *(const v8h*)p;
  v8h hi = *(const v8h*)(p + 16);
  return cat16(lo, hi);
}

__device__ __forceinline__ v8f wmma_f16(v16h a, v16h b, v8f c) {
  return __builtin_amdgcn_wmma_f32_16x16x32_f16(false, a, false, b,
                                                (short)0, c, false, false);
}

__device__ __forceinline__ float sigmoidf_(float x) {
  return 1.0f / (1.0f + __expf(-x));
}

// ---------------------------------------------------------------------------
// Grid barrier over NWG co-resident blocks (device-scope atomics)
// ---------------------------------------------------------------------------
__device__ __forceinline__ void grid_barrier(unsigned* bar, unsigned nblk) {
  __syncthreads();
  if (threadIdx.x == 0) {
    __threadfence();  // release prior global writes at agent scope
    unsigned gen  = __hip_atomic_load(&bar[1], __ATOMIC_RELAXED, __HIP_MEMORY_SCOPE_AGENT);
    unsigned prev = __hip_atomic_fetch_add(&bar[0], 1u, __ATOMIC_ACQ_REL, __HIP_MEMORY_SCOPE_AGENT);
    if (prev == nblk - 1u) {
      __hip_atomic_store(&bar[0], 0u, __ATOMIC_RELAXED, __HIP_MEMORY_SCOPE_AGENT);
      __hip_atomic_fetch_add(&bar[1], 1u, __ATOMIC_ACQ_REL, __HIP_MEMORY_SCOPE_AGENT);
    } else {
      while (__hip_atomic_load(&bar[1], __ATOMIC_ACQUIRE, __HIP_MEMORY_SCOPE_AGENT) == gen) {
        __builtin_amdgcn_s_sleep(1);
      }
    }
  }
  __syncthreads();
}

__device__ __forceinline__ const float* sel4(int g, const float* a, const float* b,
                                             const float* c, const float* d) {
  return g == 0 ? a : (g == 1 ? b : (g == 2 ? c : d));
}

// ---------------------------------------------------------------------------
// Kernel 0a: fp32 -> fp16 conversion of x
// ---------------------------------------------------------------------------
__global__ void k_cvt_x(const float* __restrict__ x, _Float16* __restrict__ x16, int n) {
  int stride = gridDim.x * blockDim.x;
  for (int i = blockIdx.x * blockDim.x + threadIdx.x; i < n; i += stride)
    x16[i] = (_Float16)x[i];
}

// ---------------------------------------------------------------------------
// Kernel 0b: pack W/U gate stacks to f16, combine biases, init h0 + barrier
// ---------------------------------------------------------------------------
__global__ void k_pack(const float* Wf, const float* Wi, const float* Wo, const float* Wc,
                       const float* bWf, const float* bWi, const float* bWo, const float* bWc,
                       const float* Uf, const float* Ui, const float* Uo, const float* Uc,
                       const float* bUf, const float* bUi, const float* bUo, const float* bUc,
                       _Float16* __restrict__ W16, _Float16* __restrict__ U16,
                       float* __restrict__ ball, _Float16* __restrict__ hbuf0,
                       unsigned* __restrict__ bar) {
  int stride = gridDim.x * blockDim.x;
  int tid0 = blockIdx.x * blockDim.x + threadIdx.x;
  // Uall: (2048, 512)
  for (int i = tid0; i < G4 * H_; i += stride) {
    int row = i >> 9, col = i & 511;
    const float* U = sel4(row >> 9, Uf, Ui, Uo, Uc);
    U16[i] = (_Float16)U[(size_t)(row & 511) * H_ + col];
  }
  // Wall: (2048, 256)
  for (int i = tid0; i < G4 * I_; i += stride) {
    int row = i >> 8, col = i & 255;
    const float* W = sel4(row >> 9, Wf, Wi, Wo, Wc);
    W16[i] = (_Float16)W[(size_t)(row & 511) * I_ + col];
  }
  // ball = bW + bU (2048)
  for (int i = tid0; i < G4; i += stride) {
    int g = i >> 9, r = i & 511;
    ball[i] = sel4(g, bWf, bWi, bWo, bWc)[r] + sel4(g, bUf, bUi, bUo, bUc)[r];
  }
  // h0 = 0
  for (int i = tid0; i < B_ * H_; i += stride) hbuf0[i] = (_Float16)0.0f;
  if (tid0 == 0) { bar[0] = 0u; bar[1] = 0u; }
}

// ---------------------------------------------------------------------------
// Kernel 1: xW = x(32768,256) * Wall^T(256,2048) + ball  -- WMMA GEMM
// Each wave computes a 16(M) x 64(N) strip; 8 waves / block.
// ---------------------------------------------------------------------------
__global__ __launch_bounds__(256) void k_gemm_xw(const _Float16* __restrict__ X16,
                                                 const _Float16* __restrict__ W16,
                                                 const float* __restrict__ ball,
                                                 float* __restrict__ xW) {
  const int lane = threadIdx.x & 31;
  const int wv   = threadIdx.x >> 5;
  const int gw   = blockIdx.x * 8 + wv;       // global wave id
  const int ntb  = gw & 31;                   // 32 N-blocks of 64
  const int mt   = gw >> 5;                   // 2048 M-tiles of 16
  if (mt >= (T_ * B_) / 16) return;

  v8f acc[4];
#pragma unroll
  for (int j = 0; j < 4; ++j)
    acc[j] = (v8f){0.f, 0.f, 0.f, 0.f, 0.f, 0.f, 0.f, 0.f};

  for (int kk = 0; kk < I_; kk += 32) {
    v16h a = load_A(X16 + (size_t)(mt * 16) * I_ + kk, I_, lane);
#pragma unroll
    for (int j = 0; j < 4; ++j) {
      v16h b = load_B(W16 + (size_t)(ntb * 64 + j * 16) * I_ + kk, I_, lane);
      acc[j] = wmma_f16(a, b, acc[j]);
    }
  }

  const int mr = (lane >> 4) << 3;            // 0 or 8
#pragma unroll
  for (int j = 0; j < 4; ++j) {
    int n = ntb * 64 + j * 16 + (lane & 15);
    float bias = ball[n];
#pragma unroll
    for (int r = 0; r < 8; ++r)
      xW[(size_t)(mt * 16 + mr + r) * G4 + n] = acc[j][r] + bias;
  }
}

// ---------------------------------------------------------------------------
// Kernel 2: persistent recurrent LSTM. 16 WGs x 256 threads (8 waves).
// WG w owns hidden slice [32w,32w+32) => 128 gate rows of Uall held in LDS.
// ---------------------------------------------------------------------------
__global__ __launch_bounds__(256) void k_lstm(const float* __restrict__ xW,
                                              const _Float16* __restrict__ U16,
                                              _Float16* __restrict__ hbuf0,
                                              _Float16* __restrict__ hbuf1,
                                              float* __restrict__ hseq,
                                              float* __restrict__ hfin,
                                              float* __restrict__ cfin,
                                              unsigned* __restrict__ bar) {
  __shared__ _Float16 sU[GSL * H_];   // 128 KB, resident for all 512 steps
  __shared__ _Float16 sH[B_ * H_];    //  64 KB, h(t) staged per step
  __shared__ float    sG[B_ * GSL];   //  32 KB, gate pre-activations

  const int w    = blockIdx.x;        // 0..15
  const int tid  = threadIdx.x;
  const int lane = tid & 31;
  const int wv   = tid >> 5;          // 0..7: owns N-tile wv (16 gate cols)

  // ---- preload this WG's 128 U rows into LDS (async) ----
  {
    int row  = tid >> 1;                               // 0..127 (local gate row)
    int half = tid & 1;                                // halves [0,256) / [256,512)
    int grow = (row >> 5) * H_ + w * HSL + (row & 31); // global Uall row
    const _Float16* src = U16 + (size_t)grow * H_ + half * 256;
    _Float16* dst = &sU[row * H_ + half * 256];
#pragma unroll
    for (int it = 0; it < 32; ++it)
      async_copy_b128(dst + it * 8, src + it * 8);
  }
  wait_async_zero();
  __syncthreads();

  // ---- per-thread recurrent state: hidden j, batch rows bg*8..bg*8+7 ----
  const int j  = tid & 31;
  const int bg = tid >> 5;
  float creg[8], hreg[8];
#pragma unroll
  for (int r = 0; r < 8; ++r) { creg[r] = 0.f; hreg[r] = 0.f; }

  const int lc   = wv * 16 + (lane & 15);                   // local gate col
  const int gcol = (lc >> 5) * H_ + w * HSL + (lc & 31);    // global gate col
  const int mr   = (lane >> 4) << 3;

  for (int t = 0; t < T_; ++t) {
    const _Float16* cur = (t & 1) ? hbuf1 : hbuf0;
    _Float16*       nxt = (t & 1) ? hbuf0 : hbuf1;

    // ---- async-stage h(t) into LDS: 64KB, 16 x b128 per thread ----
    {
      const _Float16* src = cur + tid * 128;
      _Float16* dst = &sH[tid * 128];
#pragma unroll
      for (int it = 0; it < 16; ++it)
        async_copy_b128(dst + it * 8, src + it * 8);
    }
    wait_async_zero();
    __syncthreads();

    // ---- g(64x128) = xW_t + h(64x512) @ Uslice^T : WMMA from LDS ----
    v8f acc[4];
#pragma unroll
    for (int mt = 0; mt < 4; ++mt) {
      const float* p = xW + ((size_t)(t * B_ + mt * 16 + mr)) * G4 + gcol;
#pragma unroll
      for (int r = 0; r < 8; ++r) acc[mt][r] = p[(size_t)r * G4];
    }
    for (int kk = 0; kk < H_; kk += 32) {
      v16h b = load_B(&sU[(wv * 16) * H_ + kk], H_, lane);  // reuse across M
#pragma unroll
      for (int mt = 0; mt < 4; ++mt) {
        v16h a = load_A(&sH[(mt * 16) * H_ + kk], H_, lane);
        acc[mt] = wmma_f16(a, b, acc[mt]);
      }
    }

    // ---- scatter gate pre-activations to LDS ----
    {
      int n = wv * 16 + (lane & 15);
#pragma unroll
      for (int mt = 0; mt < 4; ++mt)
#pragma unroll
        for (int r = 0; r < 8; ++r)
          sG[(mt * 16 + mr + r) * GSL + n] = acc[mt][r];
    }
    __syncthreads();

    // ---- gate combine: f,i,o,c share hidden index j ----
#pragma unroll
    for (int r = 0; r < 8; ++r) {
      int b = bg * 8 + r;
      float gf = sG[b * GSL +       j];
      float gi = sG[b * GSL +  32 + j];
      float go = sG[b * GSL +  64 + j];
      float gc = sG[b * GSL +  96 + j];
      float f  = sigmoidf_(gf);
      float ig = sigmoidf_(gi);
      float o  = sigmoidf_(go);
      float c  = f * creg[r] + ig * tanhf(gc);
      creg[r]  = c;
      float h  = o * tanhf(c);
      hreg[r]  = h;
      hseq[((size_t)t * B_ + b) * H_ + w * HSL + j] = h;
      nxt[b * H_ + w * HSL + j] = (_Float16)h;
    }

    grid_barrier(bar, NWG);   // includes __syncthreads + agent fence
  }

  // ---- final states ----
#pragma unroll
  for (int r = 0; r < 8; ++r) {
    int b = bg * 8 + r;
    hfin[b * H_ + w * HSL + j] = hreg[r];
    cfin[b * H_ + w * HSL + j] = creg[r];
  }
}

// ---------------------------------------------------------------------------
// Host launcher
// ---------------------------------------------------------------------------
extern "C" void kernel_launch(void* const* d_in, const int* in_sizes, int n_in,
                              void* d_out, int out_size, void* d_ws, size_t ws_size,
                              hipStream_t stream) {
  (void)in_sizes; (void)n_in; (void)out_size; (void)ws_size;
  const float* x   = (const float*)d_in[0];
  const float* Wf  = (const float*)d_in[1];
  const float* bWf = (const float*)d_in[2];
  const float* Wi  = (const float*)d_in[3];
  const float* bWi = (const float*)d_in[4];
  const float* Wo  = (const float*)d_in[5];
  const float* bWo = (const float*)d_in[6];
  const float* Wc  = (const float*)d_in[7];
  const float* bWc = (const float*)d_in[8];
  const float* Uf  = (const float*)d_in[9];
  const float* bUf = (const float*)d_in[10];
  const float* Ui  = (const float*)d_in[11];
  const float* bUi = (const float*)d_in[12];
  const float* Uo  = (const float*)d_in[13];
  const float* bUo = (const float*)d_in[14];
  const float* Uc  = (const float*)d_in[15];
  const float* bUc = (const float*)d_in[16];

  char* ws = (char*)d_ws;
  size_t off = 0;
  float*     xW    = (float*)(ws + off);     off += (size_t)T_ * B_ * G4 * sizeof(float);
  _Float16*  X16   = (_Float16*)(ws + off);  off += (size_t)T_ * B_ * I_ * sizeof(_Float16);
  _Float16*  W16   = (_Float16*)(ws + off);  off += (size_t)G4 * I_ * sizeof(_Float16);
  _Float16*  U16   = (_Float16*)(ws + off);  off += (size_t)G4 * H_ * sizeof(_Float16);
  float*     ball  = (float*)(ws + off);     off += (size_t)G4 * sizeof(float);
  _Float16*  hbuf0 = (_Float16*)(ws + off);  off += (size_t)B_ * H_ * sizeof(_Float16);
  _Float16*  hbuf1 = (_Float16*)(ws + off);  off += (size_t)B_ * H_ * sizeof(_Float16);
  unsigned*  bar   = (unsigned*)(ws + off);  off += 256;

  float* hseq = (float*)d_out;
  float* hfin = hseq + (size_t)T_ * B_ * H_;
  float* cfin = hfin + (size_t)B_ * H_;

  k_cvt_x<<<4096, 256, 0, stream>>>(x, X16, T_ * B_ * I_);
  k_pack<<<2048, 256, 0, stream>>>(Wf, Wi, Wo, Wc, bWf, bWi, bWo, bWc,
                                   Uf, Ui, Uo, Uc, bUf, bUi, bUo, bUc,
                                   W16, U16, ball, hbuf0, bar);
  k_gemm_xw<<<8192, 256, 0, stream>>>(X16, W16, ball, xW);
  k_lstm<<<NWG, 256, 0, stream>>>(xW, U16, hbuf0, hbuf1, hseq, hfin, cfin, bar);
}
